// CrissCrossAttention_11948599017922
// MI455X (gfx1250) — matrix operand order repaired and checked by reading, not compiled, not run on hardware
//
#include <hip/hip_runtime.h>

#define BB 8
#define CC 512
#define CQD 64
#define HH 128
#define WW 128
#define NPIX (BB * HH * WW)
#define GAMMA_V 1.0f

typedef __attribute__((ext_vector_type(16))) __bf16 v16bf;
typedef __attribute__((ext_vector_type(8))) float v8f;
typedef __attribute__((ext_vector_type(4))) int v4i;
typedef __attribute__((address_space(1))) v4i gv4i;
typedef __attribute__((address_space(3))) v4i lv4i;

union Frag { v16bf v; unsigned u[8]; };
union F4U { float4 f; float a[4]; };

// ---------------- async global->LDS (CDNA5) ------------------------------
#if defined(__has_builtin)
#if __has_builtin(__builtin_amdgcn_global_load_async_to_lds_b128)
#define HAVE_ASYNC_LDS 1
#endif
#endif
#ifndef HAVE_ASYNC_LDS
#define HAVE_ASYNC_LDS 0
#endif

__device__ __forceinline__ void copy16_g2l(void* ldsDst, const void* gSrc) {
#if HAVE_ASYNC_LDS
  __builtin_amdgcn_global_load_async_to_lds_b128(
      (gv4i*)(void*)const_cast<void*>(gSrc), (lv4i*)ldsDst, 0, 0);
#else
  *(uint4*)ldsDst = *(const uint4*)gSrc;
#endif
}

__device__ __forceinline__ void copy_wait() {
#if HAVE_ASYNC_LDS
#if __has_builtin(__builtin_amdgcn_s_wait_asynccnt)
  __builtin_amdgcn_s_wait_asynccnt(0);
#else
  asm volatile("s_wait_asynccnt 0" ::: "memory");
#endif
#endif
}

// ---------------- bf16 helpers -------------------------------------------
__device__ __forceinline__ unsigned short f2bf(float f) {
  unsigned u = __float_as_uint(f);
  u += 0x7FFFu + ((u >> 16) & 1u);  // round-to-nearest-even
  return (unsigned short)(u >> 16);
}
__device__ __forceinline__ unsigned pack2(float lo, float hi) {
  return (unsigned)f2bf(lo) | ((unsigned)f2bf(hi) << 16);
}

__device__ __forceinline__ v8f wmma_bf16(v16bf a, v16bf b, v8f c) {
  return __builtin_amdgcn_wmma_f32_16x16x32_bf16(false, a, false, b, (short)0, c,
                                                 false, false);
}

// A fragment, LDS laid out [m][k-pair] (dwords).  Per lane: two b128 loads.
// K pair dword index for VGPR j: (j>=4 ? 8 : 0) + half*4 + (j&3)
__device__ __forceinline__ v16bf fragA_row32(const unsigned* lds, int pitchU,
                                             int m0, int lane) {
  const unsigned* p = lds + (m0 + (lane & 15)) * pitchU + ((lane >> 4) & 1) * 4;
  Frag f;
#pragma unroll
  for (int j = 0; j < 4; ++j) { f.u[j] = p[j]; f.u[4 + j] = p[8 + j]; }
  return f.v;
}

// B fragment, LDS laid out [n][k-pair] (dwords).  Per lane: two b128 loads.
// K pair dword index for VGPR j: (j>=4 ? 8 : 0) + half*4 + (j&3)
__device__ __forceinline__ v16bf fragB_row32(const unsigned* lds, int pitchU,
                                             int n0, int lane) {
  const unsigned* p = lds + (n0 + (lane & 15)) * pitchU + ((lane >> 4) & 1) * 4;
  Frag f;
#pragma unroll
  for (int j = 0; j < 4; ++j) { f.u[j] = p[j]; f.u[4 + j] = p[8 + j]; }
  return f.v;
}

// ---------------------------------------------------------------------------
// Kernel 1: projection GEMM.  Block = one (b,h) row (M=128 pixels) x 64-wide
// N tile; K=C=512 chunked by 32.  Staging packs (k,k+1) bf16 pairs into
// dwords so fragment loads are contiguous ds_load_b128.
// ---------------------------------------------------------------------------
__global__ __launch_bounds__(256) void k_proj(const float* __restrict__ x,
                                              const float* __restrict__ Wm,
                                              const float* __restrict__ bias,
                                              unsigned short* __restrict__ out,
                                              int Nld) {
  __shared__ unsigned ldsA[128 * 20];  // [m=w][kpair]  pitch 80B
  __shared__ unsigned ldsB[64 * 20];   // [n][kpair]    pitch 80B
  const int bh = blockIdx.x;
  const int b = bh / HH, h = bh % HH;
  const int nbase = blockIdx.y * 64;
  const int tid = threadIdx.x, lane = tid & 31, wid = tid >> 5;
  const int m0 = wid * 16;
  const int kp = tid >> 4;              // 0..15 -> k pair (2kp, 2kp+1)
  const int wseg = (tid & 15) * 8;      // A: 8 w positions
  const int nseg = (tid & 15) * 4;      // B: 4 n positions

  v8f acc[4] = {};
#pragma unroll 1
  for (int kk = 0; kk < CC; kk += 32) {
    __syncthreads();
    {  // stage A: rows c = kk+2kp, kk+2kp+1 of x over 8 w's -> pair dwords
      const float* r0 = x + (((unsigned)b * CC + kk + 2 * kp) * HH + h) * WW + wseg;
      const float* r1 = r0 + HH * WW;
      F4U e0, e1, o0, o1;
      e0.f = ((const float4*)r0)[0]; e1.f = ((const float4*)r0)[1];
      o0.f = ((const float4*)r1)[0]; o1.f = ((const float4*)r1)[1];
#pragma unroll
      for (int i = 0; i < 4; ++i) {
        ldsA[(wseg + i) * 20 + kp]     = pack2(e0.a[i], o0.a[i]);
        ldsA[(wseg + 4 + i) * 20 + kp] = pack2(e1.a[i], o1.a[i]);
      }
    }
    {  // stage B: rows c = kk+2kp, kk+2kp+1 of Wm over 4 n's
      const float* r0 = Wm + (unsigned)(kk + 2 * kp) * Nld + nbase + nseg;
      const float* r1 = r0 + Nld;
      F4U e, o;
      e.f = *(const float4*)r0;
      o.f = *(const float4*)r1;
#pragma unroll
      for (int i = 0; i < 4; ++i)
        ldsB[(nseg + i) * 20 + kp] = pack2(e.a[i], o.a[i]);
    }
    __syncthreads();
    const v16bf fa = fragA_row32(ldsA, 20, m0, lane);
#pragma unroll
    for (int nt = 0; nt < 4; ++nt) {
      const v16bf fb = fragB_row32(ldsB, 20, nt * 16, lane);
      acc[nt] = wmma_bf16(fa, fb, acc[nt]);
    }
  }

  const int half = (lane >> 4) & 1, nl = lane & 15;
#pragma unroll
  for (int nt = 0; nt < 4; ++nt) {
    const int n = nbase + nt * 16 + nl;
    const float bv = bias[n];
#pragma unroll
    for (int r = 0; r < 8; ++r) {
      const int m = m0 + r + 8 * half;
      out[((unsigned)bh * WW + m) * Nld + n] = f2bf(acc[nt][r] + bv);
    }
  }
}

// ---------------------------------------------------------------------------
// Kernel 2: energies.  isCol=1: block=(b,w) column, E_H=Qcol.Kcol^T, -inf
// diagonal; isCol=0: block=(b,h) row, E_W.  M=N=128, K=64.  Q/K already bf16
// with K contiguous (pairs natural), staged with async b128 copies.
// ---------------------------------------------------------------------------
__global__ __launch_bounds__(256) void k_energy(const unsigned short* __restrict__ qws,
                                                const unsigned short* __restrict__ kws,
                                                float* __restrict__ E, int isCol) {
  __shared__ unsigned ldsQ[128 * 36];  // [m][kpair] pitch 144B
  __shared__ unsigned ldsK[128 * 36];  // [n][kpair]
  const int b = blockIdx.x / 128, s = blockIdx.x % 128;
  const int tid = threadIdx.x, lane = tid & 31, wid = tid >> 5;
  {
    const int row = tid >> 1, seg = (tid & 1) * 32;  // 32 bf16 = 16 dwords
    const unsigned off = isCol ? (((unsigned)(b * HH + row) * WW + s) * CQD + seg)
                               : (((unsigned)(b * HH + s) * WW + row) * CQD + seg);
    const int dbase = row * 36 + (tid & 1) * 16;
#pragma unroll
    for (int i = 0; i < 4; ++i) {
      copy16_g2l(ldsQ + dbase + i * 4, qws + off + i * 8);
      copy16_g2l(ldsK + dbase + i * 4, kws + off + i * 8);
    }
  }
  copy_wait();
  __syncthreads();

  const int m0 = wid * 16;
  v8f acc[8] = {};
#pragma unroll
  for (int kk = 0; kk < 64; kk += 32) {
    const v16bf fa = fragA_row32(ldsQ + (kk >> 1), 36, m0, lane);
#pragma unroll
    for (int nt = 0; nt < 8; ++nt) {
      const v16bf fb = fragB_row32(ldsK + (kk >> 1), 36, nt * 16, lane);
      acc[nt] = wmma_bf16(fa, fb, acc[nt]);
    }
  }

  const int half = (lane >> 4) & 1, nl = lane & 15;
#pragma unroll
  for (int nt = 0; nt < 8; ++nt) {
    const int n = nt * 16 + nl;
#pragma unroll
    for (int r = 0; r < 8; ++r) {
      const int m = m0 + r + 8 * half;
      float v = acc[nt][r];
      unsigned off;
      if (isCol) {
        if (m == n) v = -3.0e38f;  // diagonal mask (h == g)
        off = ((unsigned)(b * HH + m) * WW + s) * 256 + n;
      } else {
        off = ((unsigned)(b * HH + s) * WW + m) * 256 + 128 + n;
      }
      E[off] = v;
    }
  }
}

// ---------------------------------------------------------------------------
// Kernel 3: softmax over 256 concatenated logits, one wave per pixel.
// ---------------------------------------------------------------------------
__global__ __launch_bounds__(256) void k_softmax(const float* __restrict__ E,
                                                 unsigned short* __restrict__ A) {
  const int tid = threadIdx.x, lane = tid & 31, wid = tid >> 5;
  const unsigned pix = (unsigned)blockIdx.x * 8 + wid;
  const float* src = E + (size_t)pix * 256 + lane * 8;
  F4U a0, a1;
  a0.f = ((const float4*)src)[0];
  a1.f = ((const float4*)src)[1];
  float v[8] = {a0.a[0], a0.a[1], a0.a[2], a0.a[3],
                a1.a[0], a1.a[1], a1.a[2], a1.a[3]};
  float m = v[0];
#pragma unroll
  for (int i = 1; i < 8; ++i) m = fmaxf(m, v[i]);
#pragma unroll
  for (int off = 16; off > 0; off >>= 1) m = fmaxf(m, __shfl_xor(m, off, 32));
  float s = 0.0f;
#pragma unroll
  for (int i = 0; i < 8; ++i) { v[i] = __expf(v[i] - m); s += v[i]; }
#pragma unroll
  for (int off = 16; off > 0; off >>= 1) s += __shfl_xor(s, off, 32);
  const float inv = 1.0f / s;
  const uint4 o = make_uint4(pack2(v[0] * inv, v[1] * inv),
                             pack2(v[2] * inv, v[3] * inv),
                             pack2(v[4] * inv, v[5] * inv),
                             pack2(v[6] * inv, v[7] * inv));
  ((uint4*)(A + (size_t)pix * 256))[lane] = o;
}

// ---------------------------------------------------------------------------
// Kernel 4: output GEMM.  isCol=1: block=(b,w), out[h, c-tile] = A_H . Vcol;
// isCol=0: block=(b,h), out[w, c-tile] += A_W . Vrow.  M=128, N=128(c tile),
// K=128 (LDS-chunked by 64).  A staged async (natural pair layout); V staged
// with on-the-fly (k,k+1) pair packing so B fragments are ds_load_b128.
// ---------------------------------------------------------------------------
__global__ __launch_bounds__(256) void k_outgemm(const unsigned short* __restrict__ Att,
                                                 const unsigned short* __restrict__ vws,
                                                 float* __restrict__ accw,
                                                 int isCol, int doAcc) {
  __shared__ unsigned ldsA[128 * 68];  // [m][kpair] full K=128, pitch 272B
  __shared__ unsigned ldsV[128 * 36];  // [n=c][kpair] K chunk 64, pitch 144B
  const int b = blockIdx.x / 128, s = blockIdx.x % 128;
  const int cb = blockIdx.y * 128;
  const int tid = threadIdx.x, lane = tid & 31, wid = tid >> 5;

  {  // stage attention tile 128x128 bf16 (K contiguous -> natural pairs)
    const int row = tid >> 1, segh = (tid & 1);  // 64 bf16 halves
    const unsigned pixoff = isCol ? ((unsigned)(b * HH + row) * WW + s)
                                  : ((unsigned)(b * HH + s) * WW + row);
    const unsigned short* src = Att + (size_t)pixoff * 256 + (isCol ? 0 : 128) + segh * 64;
    unsigned* dst = ldsA + row * 68 + segh * 32;
#pragma unroll
    for (int i = 0; i < 8; ++i) copy16_g2l(dst + i * 4, src + i * 8);
  }
  copy_wait();

  const int m0 = wid * 16;
  const int kp = tid >> 3;             // 0..31 -> g pair within K chunk
  const int cseg = (tid & 7) * 16;     // 16 c positions
  v8f acc[8] = {};
#pragma unroll 1
  for (int kk = 0; kk < 128; kk += 64) {
    __syncthreads();
    {  // stage V chunk: rows g = kk+2kp, kk+2kp+1, pack pairs into [c][kpair]
      const int g0 = kk + 2 * kp;
      const unsigned pe = isCol ? ((unsigned)(b * HH + g0) * WW + s)
                                : ((unsigned)(b * HH + s) * WW + g0);
      const unsigned po = isCol ? (pe + WW) : (pe + 1);
      const uint4* se = (const uint4*)(vws + (size_t)pe * 512 + cb + cseg);
      const uint4* so = (const uint4*)(vws + (size_t)po * 512 + cb + cseg);
      const uint4 e0 = se[0], e1 = se[1], o0 = so[0], o1 = so[1];
      const unsigned ev[8] = {e0.x, e0.y, e0.z, e0.w, e1.x, e1.y, e1.z, e1.w};
      const unsigned od[8] = {o0.x, o0.y, o0.z, o0.w, o1.x, o1.y, o1.z, o1.w};
#pragma unroll
      for (int i = 0; i < 8; ++i) {
        ldsV[(cseg + 2 * i) * 36 + kp]     = (ev[i] & 0xffffu) | (od[i] << 16);
        ldsV[(cseg + 2 * i + 1) * 36 + kp] = (ev[i] >> 16) | (od[i] & 0xffff0000u);
      }
    }
    __syncthreads();
#pragma unroll
    for (int k32 = 0; k32 < 64; k32 += 32) {
      const v16bf fa = fragA_row32(ldsA + ((kk + k32) >> 1), 68, m0, lane);
#pragma unroll
      for (int nt = 0; nt < 8; ++nt) {
        const v16bf fb = fragB_row32(ldsV + (k32 >> 1), 36, nt * 16, lane);
        acc[nt] = wmma_bf16(fa, fb, acc[nt]);
      }
    }
  }

  const int half = (lane >> 4) & 1, nl = lane & 15;
#pragma unroll
  for (int nt = 0; nt < 8; ++nt) {
    const int c = cb + nt * 16 + nl;
#pragma unroll
    for (int r = 0; r < 8; ++r) {
      const int m = m0 + r + 8 * half;
      const unsigned off = isCol ? (((unsigned)(b * HH + m) * WW + s) * 512 + c)
                                 : (((unsigned)(b * HH + s) * WW + m) * 512 + c);
      const float v = acc[nt][r];
      accw[off] = doAcc ? (accw[off] + v) : v;
    }
  }
}

// ---------------------------------------------------------------------------
// Kernel 5: finalize.  out[b,c,h,w] = GAMMA * acc[b,h,w,c] + x[b,c,h,w]
// ---------------------------------------------------------------------------
__global__ __launch_bounds__(256) void k_final(const float* __restrict__ accw,
                                               const float* __restrict__ x,
                                               float* __restrict__ out) {
  __shared__ float lds[32 * 36];
  const int bh = blockIdx.x, b = bh / HH, h = bh % HH;
  const int c0 = blockIdx.y * 32, w0 = blockIdx.z * 32;
  const int tid = threadIdx.x;
  {
    const int wl = tid >> 3, cs = (tid & 7) * 4;
    const float4 v = *(const float4*)(accw + ((unsigned)bh * WW + w0 + wl) * 512 + c0 + cs);
    *(float4*)(lds + wl * 36 + cs) = v;
  }
  __syncthreads();
  {
    const int cl = tid >> 3, wseg = (tid & 7) * 4;
    const float g0 = lds[(wseg + 0) * 36 + cl];
    const float g1 = lds[(wseg + 1) * 36 + cl];
    const float g2 = lds[(wseg + 2) * 36 + cl];
    const float g3 = lds[(wseg + 3) * 36 + cl];
    const unsigned xo = (((unsigned)b * CC + c0 + cl) * HH + h) * WW + w0 + wseg;
    F4U xv; xv.f = *(const float4*)(x + xo);
    *(float4*)(out + xo) = make_float4(GAMMA_V * g0 + xv.a[0], GAMMA_V * g1 + xv.a[1],
                                       GAMMA_V * g2 + xv.a[2], GAMMA_V * g3 + xv.a[3]);
  }
}

// ---------------------------------------------------------------------------
extern "C" void kernel_launch(void* const* d_in, const int* in_sizes, int n_in,
                              void* d_out, int out_size, void* d_ws, size_t ws_size,
                              hipStream_t stream) {
  const float* x  = (const float*)d_in[0];
  const float* Wq = (const float*)d_in[1];
  const float* bq = (const float*)d_in[2];
  const float* Wk = (const float*)d_in[3];
  const float* bk = (const float*)d_in[4];
  const float* Wv = (const float*)d_in[5];
  const float* bv = (const float*)d_in[6];
  float* out = (float*)d_out;

  char* ws = (char*)d_ws;
  unsigned short* qws = (unsigned short*)ws;  ws += (size_t)NPIX * CQD * 2;   // 16.8 MB
  unsigned short* kws = (unsigned short*)ws;  ws += (size_t)NPIX * CQD * 2;   // 16.8 MB
  unsigned short* vws = (unsigned short*)ws;  ws += (size_t)NPIX * CC * 2;    // 134 MB
  float*          Ews = (float*)ws;           ws += (size_t)NPIX * 256 * 4;   // 134 MB
  unsigned short* Aws = (unsigned short*)ws;  ws += (size_t)NPIX * 256 * 2;   // 67 MB
  float*          accw = (float*)ws;                                          // 268 MB

  const dim3 blk(256);
  k_proj<<<dim3(BB * HH, 1), blk, 0, stream>>>(x, Wq, bq, qws, CQD);
  k_proj<<<dim3(BB * HH, 1), blk, 0, stream>>>(x, Wk, bk, kws, CQD);
  k_proj<<<dim3(BB * HH, 8), blk, 0, stream>>>(x, Wv, bv, vws, CC);
  k_energy<<<dim3(BB * WW), blk, 0, stream>>>(qws, kws, Ews, 1);
  k_energy<<<dim3(BB * HH), blk, 0, stream>>>(qws, kws, Ews, 0);
  k_softmax<<<dim3(NPIX / 8), blk, 0, stream>>>(Ews, Aws);
  k_outgemm<<<dim3(BB * WW, 4), blk, 0, stream>>>(Aws, vws, accw, 1, 0);
  k_outgemm<<<dim3(BB * HH, 4), blk, 0, stream>>>(Aws, vws, accw, 0, 1);
  k_final<<<dim3(BB * HH, 16, 4), blk, 0, stream>>>(accw, x, out);
}
